// GroupedQueryAttention_20976620274211
// MI455X (gfx1250) — matrix operand order repaired and checked by reading, not compiled, not run on hardware
//
#include <hip/hip_runtime.h>
#include <hip/hip_bf16.h>

// ---------------------------------------------------------------------------
// GQA attention for MI455X (gfx1250, wave32, WMMA 16x16x32 bf16, f32 accum)
// ---------------------------------------------------------------------------

#define S_LEN 4096
#define D_MODEL 1024
#define N_HEADS 16
#define N_KVH 4
#define HEAD_DIM 64
#define DKV (N_KVH * HEAD_DIM)      // 256
#define ATTN_SCALE 0.125f           // 64^-0.5

typedef __attribute__((ext_vector_type(16))) __bf16 v16bf;
typedef __attribute__((ext_vector_type(8)))  __bf16 v8bf;
typedef __attribute__((ext_vector_type(8)))  float  v8f;

static __device__ __forceinline__ v8f v8f_zero() {
    v8f r;
#pragma unroll
    for (int i = 0; i < 8; ++i) r[i] = 0.0f;
    return r;
}

static __device__ __forceinline__ v16bf cat8(v8bf lo, v8bf hi) {
    v16bf r;
#pragma unroll
    for (int i = 0; i < 8; ++i) { r[i] = lo[i]; r[i + 8] = hi[i]; }
    return r;
}

static __device__ __forceinline__ v8f wmma_bf16(v16bf a, v16bf b, v8f c) {
    // D = A(16x32 bf16) * B(32x16 bf16) + C(16x16 f32)
    return __builtin_amdgcn_wmma_f32_16x16x32_bf16(
        /*neg_a=*/false, a, /*neg_b=*/false, b,
        /*c_mod=*/(short)0, c, /*reuse_a=*/false, /*reuse_b=*/false);
}

// A-fragment (16x32, 16-bit): lanes 0-15 hold row M=lane with K={0..7,16..23},
// lanes 16-31 hold row M=lane-16 with K={8..15,24..31}  (ISA 7.12.2 table).
static __device__ __forceinline__ v16bf load_a_frag(const __bf16* X, int row0,
                                                    int k0, int ld) {
    const int lane = threadIdx.x & 31;
    const int half = lane >> 4;
    const int r    = lane & 15;
    const __bf16* p = X + (size_t)(row0 + r) * ld + k0 + half * 8;
    v8bf lo = *(const v8bf*)(p);        // K = half*8 + 0..7
    v8bf hi = *(const v8bf*)(p + 16);   // K = 16 + half*8 + 0..7
    return cat8(lo, hi);
}

// B-fragment (32x16, 16-bit): lanes 0-15 hold column N=lane with K={0..7,16..23},
// lanes 16-31 hold column N=lane-16 with K={8..15,24..31}. "W" is stored row-major
// [Ncols x ld] so a B column == one contiguous W row segment.
static __device__ __forceinline__ v16bf load_b_frag(const __bf16* W, int col0,
                                                    int k0, int ld) {
    const int lane = threadIdx.x & 31;
    const int half = lane >> 4;
    const int c    = lane & 15;
    const __bf16* p = W + (size_t)(col0 + c) * ld + k0 + half * 8;
    v8bf lo = *(const v8bf*)(p);
    v8bf hi = *(const v8bf*)(p + 16);
    return cat8(lo, hi);
}

static __device__ __forceinline__ void prefetch_a_frag(const __bf16* X,
                                                       int row0, int k0,
                                                       int ld) {
    const int lane = threadIdx.x & 31;
    const __bf16* p =
        X + (size_t)(row0 + (lane & 15)) * ld + k0 + (lane >> 4) * 8;
    __builtin_prefetch(p, 0, 3);        // global_prefetch_b8
    __builtin_prefetch(p + 16, 0, 3);
}

static __device__ __forceinline__ void prefetch_b_frag(const __bf16* W,
                                                       int col0, int k0,
                                                       int ld) {
    const int lane = threadIdx.x & 31;
    const __bf16* p =
        W + (size_t)(col0 + (lane & 15)) * ld + k0 + (lane >> 4) * 16;
    __builtin_prefetch(p, 0, 3);
}

// xor-swizzle within 32 lanes via ds_swizzle (group-of-32 mode, immediate):
// lane' = ((lane & 0x1f) | 0) ^ MASK. MASK<=8 keeps data in each 16-lane half.
template <int MASK>
static __device__ __forceinline__ float swz_xor(float v) {
    int i = __builtin_bit_cast(int, v);
    i = __builtin_amdgcn_ds_swizzle(i, (MASK << 10) | 0x1f);
    return __builtin_bit_cast(float, i);
}

// Batched 16-lane-half reductions over 8 rows at once: issue all 8 swizzles
// per step back-to-back so one dscnt wait covers the whole group.
template <int MASK>
static __device__ __forceinline__ void max_step8(float v[8]) {
    float t[8];
#pragma unroll
    for (int r = 0; r < 8; ++r) t[r] = swz_xor<MASK>(v[r]);
#pragma unroll
    for (int r = 0; r < 8; ++r) v[r] = fmaxf(v[r], t[r]);
}
template <int MASK>
static __device__ __forceinline__ void sum_step8(float v[8]) {
    float t[8];
#pragma unroll
    for (int r = 0; r < 8; ++r) t[r] = swz_xor<MASK>(v[r]);
#pragma unroll
    for (int r = 0; r < 8; ++r) v[r] += t[r];
}
static __device__ __forceinline__ void half16_max8(float v[8]) {
    max_step8<1>(v); max_step8<2>(v); max_step8<4>(v); max_step8<8>(v);
}
static __device__ __forceinline__ void half16_sum8(float v[8]) {
    sum_step8<1>(v); sum_step8<2>(v); sum_step8<4>(v); sum_step8<8>(v);
}

// ---------------------------------------------------------------------------
// fp32 -> bf16 conversion
// ---------------------------------------------------------------------------
__global__ void cvt_f32_to_bf16(const float* __restrict__ src,
                                __bf16* __restrict__ dst, int n) {
    for (int i = blockIdx.x * blockDim.x + threadIdx.x; i < n;
         i += gridDim.x * blockDim.x)
        dst[i] = (__bf16)src[i];
}

// ---------------------------------------------------------------------------
// C = X[M,K] * W[N,K]^T  (torch-Linear). 8 waves/WG, WG tile 128x64,
// each wave 32x32 via 2x2 WMMA accumulators.
// MODE 0: bf16 row-major out (ldo=N)   MODE 1: bf16 transposed out (ldo=M)
// MODE 2: f32 row-major out (ldo=N)
// ---------------------------------------------------------------------------
template <int MODE>
__global__ void __launch_bounds__(256, 1)
gemm_xwT(const __bf16* __restrict__ X, const __bf16* __restrict__ W,
         void* __restrict__ out, int K, int ldo) {
    const int lane = threadIdx.x & 31;
    // readfirstlane: wave id is wave-uniform -> scalar addressing/control.
    const int wave = __builtin_amdgcn_readfirstlane(threadIdx.x >> 5);
    const int m0 = blockIdx.x * 128 + (wave >> 1) * 32;
    const int n0 = blockIdx.y * 64 + (wave & 1) * 32;

    v8f acc[2][2];
    acc[0][0] = v8f_zero(); acc[0][1] = v8f_zero();
    acc[1][0] = v8f_zero(); acc[1][1] = v8f_zero();

    for (int k = 0; k < K; k += 32) {
        v16bf a0 = load_a_frag(X, m0,      k, K);
        v16bf a1 = load_a_frag(X, m0 + 16, k, K);
        v16bf b0 = load_b_frag(W, n0,      k, K);
        v16bf b1 = load_b_frag(W, n0 + 16, k, K);
        if (k + 32 < K) {   // hide L2 latency of next K tile (scalar branch)
            prefetch_a_frag(X, m0,      k + 32, K);
            prefetch_a_frag(X, m0 + 16, k + 32, K);
            prefetch_b_frag(W, n0,      k + 32, K);
            prefetch_b_frag(W, n0 + 16, k + 32, K);
        }
        // 4 independent WMMAs: scheduler fills D->C hazard slots
        acc[0][0] = wmma_bf16(a0, b0, acc[0][0]);
        acc[0][1] = wmma_bf16(a0, b1, acc[0][1]);
        acc[1][0] = wmma_bf16(a1, b0, acc[1][0]);
        acc[1][1] = wmma_bf16(a1, b1, acc[1][1]);
    }

    const int half = lane >> 4;
    const int nn   = lane & 15;
#pragma unroll
    for (int mi = 0; mi < 2; ++mi)
#pragma unroll
        for (int ni = 0; ni < 2; ++ni)
#pragma unroll
            for (int r = 0; r < 8; ++r) {
                const int row = m0 + mi * 16 + r + 8 * half;
                const int col = n0 + ni * 16 + nn;
                if (MODE == 0)
                    ((__bf16*)out)[(size_t)row * ldo + col] = (__bf16)acc[mi][ni][r];
                else if (MODE == 1)
                    ((__bf16*)out)[(size_t)col * ldo + row] = (__bf16)acc[mi][ni][r];
                else
                    ((float*)out)[(size_t)row * ldo + col] = acc[mi][ni][r];
            }
}

// ---------------------------------------------------------------------------
// Causal GQA flash attention. One wave per 16 query rows of one head.
// Q: [S, 1024] bf16 (head h at col h*64)  K: [S, 256] bf16  VT: [256, S] bf16
// O: [S, 1024] bf16 (concat heads)
//
// Row-sum trick: the rescale factor exp(old_max - new_max) is uniform across
// the 16 lanes holding a row, so the running softmax denominator is kept as
// an UNREDUCED per-lane partial and reduced once at the end.
// ---------------------------------------------------------------------------
__global__ void __launch_bounds__(256, 1)
gqa_attn(const __bf16* __restrict__ Q, const __bf16* __restrict__ Kc,
         const __bf16* __restrict__ VT, __bf16* __restrict__ O) {
    __shared__ __align__(16) __bf16 Plds[8][16 * 32];

    const int lane = threadIdx.x & 31;
    // wave-uniform scalar: keeps the causal loop + prefetch guard in SALU and
    // guarantees no EXEC masking around the WMMA region.
    const int wave = __builtin_amdgcn_readfirstlane(threadIdx.x >> 5);
    const int head = blockIdx.y;
    const int kvh  = head >> 2;
    const int kb0  = kvh * HEAD_DIM;               // base col in K / row in VT
    const int q0   = (blockIdx.x * 8 + wave) * 16; // uniform
    const int half = lane >> 4;
    const int nn   = lane & 15;
    const float NEG_INF = -__builtin_inff();

    // Q A-fragments (16 rows x 64 head dims -> two K=32 steps), loaded once.
    const v16bf aq0 = load_a_frag(Q, q0, head * HEAD_DIM + 0,  D_MODEL);
    const v16bf aq1 = load_a_frag(Q, q0, head * HEAD_DIM + 32, D_MODEL);

    v8f acc[4];  // output accum: 4 x (16 q rows x 16 head dims)
#pragma unroll
    for (int ni = 0; ni < 4; ++ni) acc[ni] = v8f_zero();

    float rmax[8], rsum[8];   // rsum: per-lane partial (reduced at the end)
#pragma unroll
    for (int r = 0; r < 8; ++r) { rmax[r] = NEG_INF; rsum[r] = 0.0f; }

    // Causal: process key blocks [j, j+32) while j <= q0+15 (uniform bound).
    const int jend = q0 + 16;
    for (int j = 0; j < jend; j += 32) {
        // ---- scores: Q(16x64) * K^T(64x32) -> two 16x16 tiles -------------
        v16bf bk0a = load_b_frag(Kc, j,      kb0 + 0,  DKV);
        v16bf bk1a = load_b_frag(Kc, j + 16, kb0 + 0,  DKV);
        v16bf bk0b = load_b_frag(Kc, j,      kb0 + 32, DKV);
        v16bf bk1b = load_b_frag(Kc, j + 16, kb0 + 32, DKV);
        if (j + 32 < jend) {
            prefetch_b_frag(Kc, j + 32, kb0, DKV);
            prefetch_b_frag(Kc, j + 48, kb0, DKV);
            prefetch_b_frag(VT, kb0, j + 32, S_LEN);
        }
        // independent pair first, then the dependent accumulating pair
        v8f s0 = wmma_bf16(aq0, bk0a, v8f_zero());
        v8f s1 = wmma_bf16(aq0, bk1a, v8f_zero());
        s0     = wmma_bf16(aq1, bk0b, s0);
        s1     = wmma_bf16(aq1, bk1b, s1);

        // ---- scale + causal mask (in-place) -------------------------------
        float bm[8];
#pragma unroll
        for (int r = 0; r < 8; ++r) {
            const int m = q0 + r + 8 * half;       // absolute query row
            float x0 = s0[r] * ATTN_SCALE;
            float x1 = s1[r] * ATTN_SCALE;
            if (j + nn > m)      x0 = NEG_INF;
            if (j + 16 + nn > m) x1 = NEG_INF;
            s0[r] = x0;
            s1[r] = x1;
            bm[r] = fmaxf(x0, x1);
        }
        half16_max8(bm);                            // batched ds_swizzle tree

        // ---- online softmax update ----------------------------------------
        float fscale[8];
#pragma unroll
        for (int r = 0; r < 8; ++r) {
            const float nm = fmaxf(rmax[r], bm[r]); // finite: j <= q0 <= m
            fscale[r] = __expf(rmax[r] - nm);
            rmax[r]   = nm;
            const float p0 = __expf(s0[r] - nm);
            const float p1 = __expf(s1[r] - nm);
            rsum[r] = rsum[r] * fscale[r] + p0 + p1;
            const int mrel = r + 8 * half;
            Plds[wave][mrel * 32 + nn]      = (__bf16)p0;
            Plds[wave][mrel * 32 + 16 + nn] = (__bf16)p1;
        }
#pragma unroll
        for (int ni = 0; ni < 4; ++ni)
#pragma unroll
            for (int r = 0; r < 8; ++r) acc[ni][r] *= fscale[r];

        // ---- P(16x32) through LDS (C-layout -> A-layout), then P*V --------
        const v16bf ap = load_a_frag(&Plds[wave][0], 0, 0, 32);
#pragma unroll
        for (int ni = 0; ni < 4; ++ni) {
            v16bf bv = load_b_frag(VT, kb0 + ni * 16, j, S_LEN);
            acc[ni] = wmma_bf16(ap, bv, acc[ni]);   // 4 independent chains
        }
    }

    // ---- final denominator reduction, normalize, store --------------------
    half16_sum8(rsum);
    float inv[8];
#pragma unroll
    for (int r = 0; r < 8; ++r) inv[r] = 1.0f / rsum[r];
#pragma unroll
    for (int ni = 0; ni < 4; ++ni)
#pragma unroll
        for (int r = 0; r < 8; ++r) {
            const int row = q0 + r + 8 * half;
            const int col = head * HEAD_DIM + ni * 16 + nn;
            O[(size_t)row * D_MODEL + col] = (__bf16)(acc[ni][r] * inv[r]);
        }
}

// ---------------------------------------------------------------------------
// Host launch
// ---------------------------------------------------------------------------
extern "C" void kernel_launch(void* const* d_in, const int* in_sizes, int n_in,
                              void* d_out, int out_size, void* d_ws,
                              size_t ws_size, hipStream_t stream) {
    (void)in_sizes; (void)n_in; (void)out_size; (void)ws_size;
    const float* x  = (const float*)d_in[0];
    const float* Wq = (const float*)d_in[1];
    const float* Wk = (const float*)d_in[2];
    const float* Wv = (const float*)d_in[3];
    const float* Wo = (const float*)d_in[4];
    float* out = (float*)d_out;

    char* ws = (char*)d_ws;
    size_t off = 0;
    auto carve = [&](size_t bytes) -> void* {
        void* p = ws + off;
        off = (off + bytes + 255) & ~(size_t)255;
        return p;
    };
    __bf16* xb  = (__bf16*)carve((size_t)S_LEN * D_MODEL * 2);  // x bf16
    __bf16* wqb = (__bf16*)carve((size_t)D_MODEL * D_MODEL * 2);
    __bf16* wkb = (__bf16*)carve((size_t)DKV * D_MODEL * 2);
    __bf16* wvb = (__bf16*)carve((size_t)DKV * D_MODEL * 2);
    __bf16* wob = (__bf16*)carve((size_t)D_MODEL * D_MODEL * 2);
    __bf16* qb  = (__bf16*)carve((size_t)S_LEN * D_MODEL * 2);  // Q [S,1024]
    __bf16* kb  = (__bf16*)carve((size_t)S_LEN * DKV * 2);      // K [S,256]
    __bf16* vtb = (__bf16*)carve((size_t)DKV * S_LEN * 2);      // V^T [256,S]
    __bf16* aob = (__bf16*)carve((size_t)S_LEN * D_MODEL * 2);  // attn out

    const dim3 blk(256);
    // fp32 -> bf16
    cvt_f32_to_bf16<<<1024, blk, 0, stream>>>(x,  xb,  S_LEN * D_MODEL);
    cvt_f32_to_bf16<<<1024, blk, 0, stream>>>(Wq, wqb, D_MODEL * D_MODEL);
    cvt_f32_to_bf16<<<256,  blk, 0, stream>>>(Wk, wkb, DKV * D_MODEL);
    cvt_f32_to_bf16<<<256,  blk, 0, stream>>>(Wv, wvb, DKV * D_MODEL);
    cvt_f32_to_bf16<<<1024, blk, 0, stream>>>(Wo, wob, D_MODEL * D_MODEL);

    // projections
    gemm_xwT<0><<<dim3(S_LEN / 128, D_MODEL / 64), blk, 0, stream>>>(
        xb, wqb, qb, D_MODEL, D_MODEL);
    gemm_xwT<0><<<dim3(S_LEN / 128, DKV / 64), blk, 0, stream>>>(
        xb, wkb, kb, D_MODEL, DKV);
    gemm_xwT<1><<<dim3(S_LEN / 128, DKV / 64), blk, 0, stream>>>(
        xb, wvb, vtb, D_MODEL, S_LEN);          // store V transposed

    // causal GQA attention
    gqa_attn<<<dim3(S_LEN / 128, N_HEADS), blk, 0, stream>>>(qb, kb, vtb, aob);

    // output projection (f32 out)
    gemm_xwT<2><<<dim3(S_LEN / 128, D_MODEL / 64), blk, 0, stream>>>(
        aob, wob, out, D_MODEL, D_MODEL);
}
